// Chronos2Attention_40810779247248
// MI455X (gfx1250) — matrix operand order repaired and checked by reading, not compiled, not run on hardware
//
#include <hip/hip_runtime.h>
#include <hip/hip_bf16.h>

// ---------------- problem constants ----------------
#define S_LEN 2048
#define BATCH 4
#define DM    1024
#define NH    16
#define DH    64
#define BS    (BATCH * S_LEN)   // 8192 tokens

typedef __attribute__((ext_vector_type(16))) __bf16          v16bf;
typedef __attribute__((ext_vector_type(8)))  float           v8f;
typedef __attribute__((ext_vector_type(4)))  float           v4f;
typedef __attribute__((ext_vector_type(8)))  unsigned short  v8us;
typedef __attribute__((ext_vector_type(4)))  unsigned short  v4us;

union BFrag { v16bf v; v8us h[2]; unsigned short u[16]; };

__device__ __forceinline__ unsigned short f2bf(float f) {
    union { __bf16 h; unsigned short u; } c;
    c.h = (__bf16)f;                       // hardware cvt, RNE
    return c.u;
}

__device__ __forceinline__ v8f wmma_bf16(const BFrag& a, const BFrag& b, v8f c) {
    return __builtin_amdgcn_wmma_f32_16x16x32_bf16(
        false, a.v, false, b.v, (short)0, c, false, false);
}

// =====================================================================
// Kernel 0: one-time weight transpose+convert: WT[n][k] = bf16(W[k][n])
// =====================================================================
__global__ __launch_bounds__(256)
void transpose_w_kernel(const float* __restrict__ Wq, const float* __restrict__ Wk,
                        const float* __restrict__ Wv, const float* __restrict__ Wo,
                        unsigned short* __restrict__ WTbase)
{
    __shared__ unsigned short t[16][17];
    const float* W = (blockIdx.z == 0) ? Wq : (blockIdx.z == 1) ? Wk
                   : (blockIdx.z == 2) ? Wv : Wo;
    unsigned short* WT = WTbase + (size_t)blockIdx.z * DM * DM;
    int k0 = blockIdx.x * 16, n0 = blockIdx.y * 16;
    t[threadIdx.y][threadIdx.x] =
        f2bf(W[(size_t)(k0 + threadIdx.y) * DM + n0 + threadIdx.x]);
    __syncthreads();
    WT[(size_t)(n0 + threadIdx.y) * DM + k0 + threadIdx.x] = t[threadIdx.x][threadIdx.y];
}

// =====================================================================
// Kernel 1: Y = X @ W (W transposed bf16), optional RoPE.
// mode==1: RoPE + store [B,H,S,Dh] (Q,K).  mode==0: store V as [B,H,Dh,S].
// grid: (BS/128, NH), block 128 (4 waves); wave w owns rows w*32..+31
// =====================================================================
__global__ __launch_bounds__(128)
void qkv_rope_kernel(const float* __restrict__ X, const unsigned short* __restrict__ WT,
                     const int* __restrict__ positions,
                     unsigned short* __restrict__ out, int mode)
{
    __shared__ alignas(16) unsigned short Xs[128][64];   // 16 KB (reused as epilogue)
    __shared__ alignas(16) unsigned short Ws[64][64];    //  8 KB

    const int tid  = threadIdx.x;
    const int wave = tid >> 5;
    const int lane = tid & 31;
    const int g    = lane >> 4;
    const int ln   = lane & 15;
    const int row0 = blockIdx.x * 128;         // global token tile base
    const int h    = blockIdx.y;
    const int col0 = h * DH;

    v8f acc[2][4];
    #pragma unroll
    for (int mt = 0; mt < 2; ++mt)
        #pragma unroll
        for (int t = 0; t < 4; ++t)
            #pragma unroll
            for (int i = 0; i < 8; ++i) acc[mt][t][i] = 0.0f;

    for (int k0 = 0; k0 < DM; k0 += 64) {
        // stage X tile 128x64 (f32 -> bf16), vectorized
        for (int j = tid; j < 128 * 16; j += 128) {
            int r = j >> 4, c4 = (j & 15) * 4;
            v4f xv = *(const v4f*)&X[(size_t)(row0 + r) * DM + k0 + c4];
            v4us o;
            o[0] = f2bf(xv[0]); o[1] = f2bf(xv[1]);
            o[2] = f2bf(xv[2]); o[3] = f2bf(xv[3]);
            *(v4us*)&Xs[r][c4] = o;
        }
        // stage WT tile 64(n) x 64(k), flat b128 copies
        for (int j = tid; j < 64 * 8; j += 128) {
            int r = j >> 3, c8 = (j & 7) * 8;
            *(v8us*)&Ws[r][c8] = *(const v8us*)&WT[(size_t)(col0 + r) * DM + k0 + c8];
        }
        __syncthreads();

        BFrag a[2][2];
        #pragma unroll
        for (int mt = 0; mt < 2; ++mt) {
            const int m = wave * 32 + mt * 16 + ln;
            #pragma unroll
            for (int c = 0; c < 2; ++c) {
                a[mt][c].h[0] = *(const v8us*)&Xs[m][c * 32 + g * 8];
                a[mt][c].h[1] = *(const v8us*)&Xs[m][c * 32 + 16 + g * 8];
            }
        }
        #pragma unroll
        for (int t = 0; t < 4; ++t) {
            #pragma unroll
            for (int c = 0; c < 2; ++c) {
                BFrag bfr;                       // shared by both m-tiles
                bfr.h[0] = *(const v8us*)&Ws[t * 16 + ln][c * 32 + g * 8];
                bfr.h[1] = *(const v8us*)&Ws[t * 16 + ln][c * 32 + 16 + g * 8];
                acc[0][t] = wmma_bf16(a[0][c], bfr, acc[0][t]);
                acc[1][t] = wmma_bf16(a[1][c], bfr, acc[1][t]);
            }
        }
        __syncthreads();
    }

    const int b  = row0 >> 11;            // /S_LEN
    const int s0 = row0 & (S_LEN - 1);

    if (mode) {
        // RoPE: (mt, tile t, reg r, lane) -> (srow = wave*32+mt*16+r+8g, d = t*16+ln)
        #pragma unroll
        for (int mt = 0; mt < 2; ++mt)
            #pragma unroll
            for (int p = 0; p < 2; ++p) {
                #pragma unroll
                for (int r = 0; r < 8; ++r) {
                    int s   = s0 + wave * 32 + mt * 16 + r + 8 * g;
                    float pos = (float)positions[s];
                    int   d0  = p * 16 + ln;                       // 0..31
                    float inv = __powf(10000.0f, -(float)d0 / 32.0f);
                    float ang = pos * inv;
                    float c = __cosf(ang), sn = __sinf(ang);
                    float x0 = acc[mt][p][r], x2 = acc[mt][p + 2][r];
                    acc[mt][p][r]     = x0 * c - x2 * sn;
                    acc[mt][p + 2][r] = x2 * c + x0 * sn;
                }
            }
        // stage row-major [srow][d] in Xs, then flat contiguous b128 stores
        #pragma unroll
        for (int mt = 0; mt < 2; ++mt)
            #pragma unroll
            for (int t = 0; t < 4; ++t)
                #pragma unroll
                for (int r = 0; r < 8; ++r)
                    Xs[wave * 32 + mt * 16 + r + 8 * g][t * 16 + ln] = f2bf(acc[mt][t][r]);
        __syncthreads();
        unsigned short* dst = out + (((size_t)b * NH + h) * S_LEN + s0) * DH;
        for (int j = tid; j < 128 * 8; j += 128)
            *(v8us*)&dst[j * 8] = *(const v8us*)&Xs[0][j * 8];
    } else {
        // V: stage transposed [d][srow] (reuse Xs as [64][128]), store [B,H,Dh,S]
        unsigned short (*Tv)[128] = (unsigned short (*)[128])&Xs[0][0];
        #pragma unroll
        for (int mt = 0; mt < 2; ++mt)
            #pragma unroll
            for (int t = 0; t < 4; ++t)
                #pragma unroll
                for (int r = 0; r < 8; ++r)
                    Tv[t * 16 + ln][wave * 32 + mt * 16 + r + 8 * g] = f2bf(acc[mt][t][r]);
        __syncthreads();
        for (int j = tid; j < 64 * 16; j += 128) {
            int d = j >> 4, c8 = (j & 15) * 8;
            *(v8us*)&out[(((size_t)b * NH + h) * DH + d) * S_LEN + s0 + c8] =
                *(const v8us*)&Tv[d][c8];
        }
    }
}

// =====================================================================
// Kernel 2: flash attention, 128-query tiles, 64-key chunks.
// Q,K bf16 [B,H,S,Dh]; V bf16 [B,H,Dh,S]; O bf16 [B,S,H*Dh]
// grid: (S/128, NH, BATCH), block 128 (4 waves, 32 queries each)
// =====================================================================
__global__ __launch_bounds__(128)
void flash_attn_kernel(const unsigned short* __restrict__ Q,
                       const unsigned short* __restrict__ K,
                       const unsigned short* __restrict__ VT,
                       unsigned short* __restrict__ O)
{
    __shared__ alignas(16) unsigned short Ks [64][64];    // [key][dh]    8 KB
    __shared__ alignas(16) unsigned short VTs[64][64];    // [dh][key]    8 KB
    __shared__ alignas(16) unsigned short Ps [4][32][64]; // per-wave P  16 KB
    __shared__ alignas(16) unsigned short Os [128][64];   // epilogue    16 KB

    const int tid  = threadIdx.x;
    const int wave = tid >> 5;
    const int lane = tid & 31;
    const int g    = lane >> 4;
    const int ln   = lane & 15;
    const int q0   = blockIdx.x * 128;
    const int h    = blockIdx.y;
    const int b    = blockIdx.z;
    const size_t bh = ((size_t)b * NH + h) * (size_t)S_LEN * DH;

    // Q A-fragments (2 m-tiles x 2 dh chunks), held for the whole pass
    BFrag qa[2][2];
    #pragma unroll
    for (int mt = 0; mt < 2; ++mt) {
        const unsigned short* qrow =
            Q + bh + (size_t)(q0 + wave * 32 + mt * 16 + ln) * DH;
        #pragma unroll
        for (int c = 0; c < 2; ++c) {
            qa[mt][c].h[0] = *(const v8us*)&qrow[c * 32 + g * 8];
            qa[mt][c].h[1] = *(const v8us*)&qrow[c * 32 + 16 + g * 8];
        }
    }

    v8f ov[2][4];
    #pragma unroll
    for (int mt = 0; mt < 2; ++mt)
        #pragma unroll
        for (int t = 0; t < 4; ++t)
            #pragma unroll
            for (int i = 0; i < 8; ++i) ov[mt][t][i] = 0.0f;

    float run_m[2][8], run_l[2][8];
    #pragma unroll
    for (int mt = 0; mt < 2; ++mt)
        #pragma unroll
        for (int r = 0; r < 8; ++r) { run_m[mt][r] = -1e30f; run_l[mt][r] = 0.0f; }

    for (int kk0 = 0; kk0 < S_LEN; kk0 += 64) {
        // stage K rows (flat 64x64) and VT rows (64 keys each)
        {
            const unsigned short* ksrc = K + bh + (size_t)kk0 * DH;
            for (int j = tid; j < 64 * 8; j += 128)
                *(v8us*)&Ks[0][j * 8] = *(const v8us*)&ksrc[j * 8];
            for (int j = tid; j < 64 * 8; j += 128) {
                int d = j >> 3, c8 = (j & 7) * 8;
                *(v8us*)&VTs[d][c8] =
                    *(const v8us*)&VT[bh + (size_t)d * S_LEN + kk0 + c8];
            }
        }
        __syncthreads();

        // prefetch next chunk while computing this one
        if (kk0 + 64 < S_LEN) {
            const char* kp = (const char*)(K + bh + (size_t)(kk0 + 64) * DH);
            __builtin_prefetch(kp + tid * 64, 0, 1);
            const char* vp = (const char*)(VT + bh + (size_t)(tid >> 1) * S_LEN
                                           + kk0 + 64 + (tid & 1) * 32);
            __builtin_prefetch(vp, 0, 1);
        }

        // scores: 2 m-tiles x 4 key subtiles; each kb feeds both m-tiles
        v8f st[2][4];
        #pragma unroll
        for (int mt = 0; mt < 2; ++mt)
            #pragma unroll
            for (int kt = 0; kt < 4; ++kt)
                #pragma unroll
                for (int i = 0; i < 8; ++i) st[mt][kt][i] = 0.0f;
        #pragma unroll
        for (int kt = 0; kt < 4; ++kt) {
            #pragma unroll
            for (int c = 0; c < 2; ++c) {
                BFrag kb;
                kb.h[0] = *(const v8us*)&Ks[kt * 16 + ln][c * 32 + g * 8];
                kb.h[1] = *(const v8us*)&Ks[kt * 16 + ln][c * 32 + 16 + g * 8];
                st[0][kt] = wmma_bf16(qa[0][c], kb, st[0][kt]);
                st[1][kt] = wmma_bf16(qa[1][c], kb, st[1][kt]);
            }
        }

        // online softmax over 64 keys for both m-tiles
        #pragma unroll
        for (int mt = 0; mt < 2; ++mt) {
            #pragma unroll
            for (int r = 0; r < 8; ++r) {
                float mx = fmaxf(fmaxf(st[mt][0][r], st[mt][1][r]),
                                 fmaxf(st[mt][2][r], st[mt][3][r]));
                #pragma unroll
                for (int off = 8; off >= 1; off >>= 1)
                    mx = fmaxf(mx, __shfl_xor(mx, off, 32));
                float nm = fmaxf(run_m[mt][r], mx);
                float corr = __expf(run_m[mt][r] - nm);
                run_m[mt][r] = nm;
                float p[4], ps = 0.0f;
                #pragma unroll
                for (int kt = 0; kt < 4; ++kt) {
                    p[kt] = __expf(st[mt][kt][r] - nm);
                    ps += p[kt];
                }
                #pragma unroll
                for (int off = 8; off >= 1; off >>= 1)
                    ps += __shfl_xor(ps, off, 32);
                run_l[mt][r] = run_l[mt][r] * corr + ps;
                #pragma unroll
                for (int t = 0; t < 4; ++t) ov[mt][t][r] *= corr;
                int m2 = mt * 16 + r + 8 * g;
                #pragma unroll
                for (int kt = 0; kt < 4; ++kt)
                    Ps[wave][m2][kt * 16 + ln] = f2bf(p[kt]);
            }
        }
        __syncthreads();

        // P @ V: A = P[32q x 64k] (2x2 frags), each vb feeds both m-tiles
        BFrag pa[2][2];
        #pragma unroll
        for (int mt = 0; mt < 2; ++mt)
            #pragma unroll
            for (int c = 0; c < 2; ++c) {
                pa[mt][c].h[0] = *(const v8us*)&Ps[wave][mt * 16 + ln][c * 32 + g * 8];
                pa[mt][c].h[1] = *(const v8us*)&Ps[wave][mt * 16 + ln][c * 32 + 16 + g * 8];
            }
        #pragma unroll
        for (int t = 0; t < 4; ++t) {
            #pragma unroll
            for (int c = 0; c < 2; ++c) {
                BFrag vb;
                vb.h[0] = *(const v8us*)&VTs[t * 16 + ln][c * 32 + g * 8];
                vb.h[1] = *(const v8us*)&VTs[t * 16 + ln][c * 32 + 16 + g * 8];
                ov[0][t] = wmma_bf16(pa[0][c], vb, ov[0][t]);
                ov[1][t] = wmma_bf16(pa[1][c], vb, ov[1][t]);
            }
        }
        __syncthreads();
    }

    // normalize, stage, contiguous row stores to [B,S,H*Dh]
    #pragma unroll
    for (int mt = 0; mt < 2; ++mt)
        #pragma unroll
        for (int t = 0; t < 4; ++t)
            #pragma unroll
            for (int r = 0; r < 8; ++r)
                Os[wave * 32 + mt * 16 + r + 8 * g][t * 16 + ln] =
                    f2bf(ov[mt][t][r] / run_l[mt][r]);
    __syncthreads();
    for (int j = tid; j < 128 * 8; j += 128) {
        int srow = j >> 3, c8 = (j & 7) * 8;
        size_t idx = ((size_t)(b * S_LEN + q0 + srow)) * (NH * DH) + h * DH + c8;
        *(v8us*)&O[idx] = *(const v8us*)&Os[srow][c8];
    }
}

// =====================================================================
// Kernel 3: out = Attn(bf16)[BS x 1024] @ Wo^T(bf16), f32 out
// grid: (BS/128, DM/64), block 128 (4 waves, 32 rows each)
// =====================================================================
__global__ __launch_bounds__(128)
void out_proj_kernel(const unsigned short* __restrict__ A,
                     const unsigned short* __restrict__ WT, float* __restrict__ out)
{
    __shared__ alignas(16) unsigned short As[128][64];
    __shared__ alignas(16) unsigned short Ws[64][64];

    const int tid  = threadIdx.x;
    const int wave = tid >> 5;
    const int lane = tid & 31;
    const int g    = lane >> 4;
    const int ln   = lane & 15;
    const int row0 = blockIdx.x * 128;
    const int col0 = blockIdx.y * 64;

    v8f acc[2][4];
    #pragma unroll
    for (int mt = 0; mt < 2; ++mt)
        #pragma unroll
        for (int t = 0; t < 4; ++t)
            #pragma unroll
            for (int i = 0; i < 8; ++i) acc[mt][t][i] = 0.0f;

    for (int k0 = 0; k0 < DM; k0 += 64) {
        for (int j = tid; j < 128 * 8; j += 128) {
            int r = j >> 3, c8 = (j & 7) * 8;
            *(v8us*)&As[r][c8] = *(const v8us*)&A[(size_t)(row0 + r) * DM + k0 + c8];
        }
        for (int j = tid; j < 64 * 8; j += 128) {
            int r = j >> 3, c8 = (j & 7) * 8;
            *(v8us*)&Ws[r][c8] = *(const v8us*)&WT[(size_t)(col0 + r) * DM + k0 + c8];
        }
        __syncthreads();

        BFrag a[2][2];
        #pragma unroll
        for (int mt = 0; mt < 2; ++mt) {
            const int m = wave * 32 + mt * 16 + ln;
            #pragma unroll
            for (int c = 0; c < 2; ++c) {
                a[mt][c].h[0] = *(const v8us*)&As[m][c * 32 + g * 8];
                a[mt][c].h[1] = *(const v8us*)&As[m][c * 32 + 16 + g * 8];
            }
        }
        #pragma unroll
        for (int t = 0; t < 4; ++t) {
            #pragma unroll
            for (int c = 0; c < 2; ++c) {
                BFrag bfr;
                bfr.h[0] = *(const v8us*)&Ws[t * 16 + ln][c * 32 + g * 8];
                bfr.h[1] = *(const v8us*)&Ws[t * 16 + ln][c * 32 + 16 + g * 8];
                acc[0][t] = wmma_bf16(a[0][c], bfr, acc[0][t]);
                acc[1][t] = wmma_bf16(a[1][c], bfr, acc[1][t]);
            }
        }
        __syncthreads();
    }

    #pragma unroll
    for (int mt = 0; mt < 2; ++mt)
        #pragma unroll
        for (int t = 0; t < 4; ++t)
            #pragma unroll
            for (int r = 0; r < 8; ++r) {
                int row = row0 + wave * 32 + mt * 16 + r + 8 * g;
                out[(size_t)row * DM + col0 + t * 16 + ln] = acc[mt][t][r];
            }
}

// =====================================================================
extern "C" void kernel_launch(void* const* d_in, const int* in_sizes, int n_in,
                              void* d_out, int out_size, void* d_ws, size_t ws_size,
                              hipStream_t stream) {
    (void)in_sizes; (void)n_in; (void)out_size; (void)ws_size;
    const int*   positions = (const int*)  d_in[0];
    const float* hidden    = (const float*)d_in[1];
    const float* wq        = (const float*)d_in[2];
    const float* wk        = (const float*)d_in[3];
    const float* wv        = (const float*)d_in[4];
    const float* wo        = (const float*)d_in[5];
    float*       out       = (float*)d_out;

    const size_t WSZ = (size_t)DM * DM;                   // 1M elems per weight
    const size_t QKV = (size_t)BATCH * NH * S_LEN * DH;   // 8,388,608 elems

    unsigned short* wt  = (unsigned short*)d_ws;          // 4 transposed weights
    unsigned short* qws = wt + 4 * WSZ;
    unsigned short* kws = qws + QKV;
    unsigned short* vws = kws + QKV;                      // stored [B,H,Dh,S]
    unsigned short* aws = vws + QKV;

    transpose_w_kernel<<<dim3(64, 64, 4), dim3(16, 16), 0, stream>>>(wq, wk, wv, wo, wt);

    dim3 blk(128);
    dim3 grid_qkv(BS / 128, NH);
    qkv_rope_kernel<<<grid_qkv, blk, 0, stream>>>(hidden, wt + 0 * WSZ, positions, qws, 1);
    qkv_rope_kernel<<<grid_qkv, blk, 0, stream>>>(hidden, wt + 1 * WSZ, positions, kws, 1);
    qkv_rope_kernel<<<grid_qkv, blk, 0, stream>>>(hidden, wt + 2 * WSZ, positions, vws, 0);

    dim3 grid_fa(S_LEN / 128, NH, BATCH);
    flash_attn_kernel<<<grid_fa, blk, 0, stream>>>(qws, kws, vws, aws);

    dim3 grid_op(BS / 128, DM / 64);
    out_proj_kernel<<<grid_op, blk, 0, stream>>>(aws, wt + 3 * WSZ, out);
}
